// GIN_81819126989475
// MI455X (gfx1250) — matrix operand order repaired and checked by reading, not compiled, not run on hardware
//
#include <hip/hip_runtime.h>

typedef __attribute__((ext_vector_type(16))) __bf16         v16bf;
typedef __attribute__((ext_vector_type(16))) unsigned short v16u;
typedef __attribute__((ext_vector_type(8)))  float          v8f;

#define HIDC  128
#define LDSTR 136   // padded ushort stride (272B) to break LDS bank conflicts

// native f32 -> bf16 (RNE in hardware; lowers to v_cvt_pk_bf16_f32 class ops)
__device__ __forceinline__ unsigned short f2bf(float f) {
  union { __bf16 h; unsigned short s; } u;
  u.h = (__bf16)f;
  return u.s;
}

union FragAB { v16u u; v16bf b; };

// ---------------------------------------------------------------------------
// Fused GIN MLP: Out = relu(In @ Wa + ba) @ Wb + bb      (In/Out: [nrows,128])
// WaT/WbT are bf16 bit patterns, pre-transposed: WT[n][k] = W[k][n].
// Block = 128 threads (4 waves); wave w owns rows [16*(4*bid+w), +16).
// ---------------------------------------------------------------------------
__global__ __launch_bounds__(128) void mlp_kernel(
    const float*          __restrict__ In,
    const unsigned short* __restrict__ WaT,
    const float*          __restrict__ Ba,
    const unsigned short* __restrict__ WbT,
    const float*          __restrict__ Bb,
    float*                __restrict__ Out,
    int nrows)
{
  __shared__ unsigned short sW[HIDC * LDSTR];        // 34816 B (Wa, then Wb)
  __shared__ unsigned short sMid[4][16 * LDSTR];     // 17408 B bf16 intermediate

  const int tid   = threadIdx.x;
  const int wave  = tid >> 5;
  const int lane  = tid & 31;
  const int lhalf = lane >> 4;   // 0|1 : lane half selects K sub-range
  const int lmod  = lane & 15;

  // stage WaT -> LDS (uint-granularity, coalesced)
  for (int i = tid; i < HIDC * (HIDC / 2); i += 128) {
    int r = i >> 6, c = i & 63;
    ((unsigned int*)(sW + r * LDSTR))[c] = ((const unsigned int*)(WaT + r * HIDC))[c];
  }
  __syncthreads();

  const int  rowBase = (blockIdx.x * 4 + wave) * 16;
  const bool valid   = rowBase < nrows;
  const int  mrow    = min(rowBase + lmod, nrows - 1);   // clamped for tail strips

  // ---------------- GEMM1: A from global (fp32 -> bf16), B from LDS ----------
  v8f acc[8] = {};
  #pragma unroll
  for (int kk = 0; kk < 4; ++kk) {
    const int kbA = kk * 32 + 8 * lhalf;               // A lane K-base
    const float* rp = In + (size_t)mrow * HIDC + kbA;
    float4 f0 = *(const float4*)(rp);
    float4 f1 = *(const float4*)(rp + 4);
    float4 f2 = *(const float4*)(rp + 16);
    float4 f3 = *(const float4*)(rp + 20);
    FragAB a;
    a.u[0]  = f2bf(f0.x); a.u[1]  = f2bf(f0.y); a.u[2]  = f2bf(f0.z); a.u[3]  = f2bf(f0.w);
    a.u[4]  = f2bf(f1.x); a.u[5]  = f2bf(f1.y); a.u[6]  = f2bf(f1.z); a.u[7]  = f2bf(f1.w);
    a.u[8]  = f2bf(f2.x); a.u[9]  = f2bf(f2.y); a.u[10] = f2bf(f2.z); a.u[11] = f2bf(f2.w);
    a.u[12] = f2bf(f3.x); a.u[13] = f2bf(f3.y); a.u[14] = f2bf(f3.z); a.u[15] = f2bf(f3.w);

    const int kbB = kk * 32 + 8 * lhalf;               // B lane K-base
    #pragma unroll
    for (int nt = 0; nt < 8; ++nt) {
      const unsigned short* bp = sW + (nt * 16 + lmod) * LDSTR + kbB;
      FragAB b;
      #pragma unroll
      for (int j = 0; j < 16; ++j) b.u[j] = bp[j < 8 ? j : j + 8];     // K = kbB + j, then kbB + 16 + (j-8)
      acc[nt] = __builtin_amdgcn_wmma_f32_16x16x32_bf16(
          false, a.b, false, b.b, (short)0, acc[nt], false, false);
    }
  }

  // bias + ReLU -> bf16 intermediate in LDS (C layout: M = r + 8*lhalf, N = lmod)
  #pragma unroll
  for (int nt = 0; nt < 8; ++nt) {
    float bias = Ba[nt * 16 + lmod];
    #pragma unroll
    for (int r = 0; r < 8; ++r) {
      float v = fmaxf(acc[nt][r] + bias, 0.0f);
      sMid[wave][(r + 8 * lhalf) * LDSTR + nt * 16 + lmod] = f2bf(v);
    }
  }
  __syncthreads();

  // stage WbT -> LDS (reuse sW)
  for (int i = tid; i < HIDC * (HIDC / 2); i += 128) {
    int r = i >> 6, c = i & 63;
    ((unsigned int*)(sW + r * LDSTR))[c] = ((const unsigned int*)(WbT + r * HIDC))[c];
  }
  __syncthreads();

  // ---------------- GEMM2: A from LDS intermediate, B from LDS ---------------
  v8f acc2[8] = {};
  #pragma unroll
  for (int kk = 0; kk < 4; ++kk) {
    const int kbA = kk * 32 + 8 * lhalf;
    const unsigned short* mp = &sMid[wave][lmod * LDSTR + kbA];
    FragAB a;
    #pragma unroll
    for (int j = 0; j < 8; ++j)  a.u[j] = mp[j];       // K = kbA + j
    #pragma unroll
    for (int j = 8; j < 16; ++j) a.u[j] = mp[8 + j];   // K = kbA + 16 + (j-8)

    const int kbB = kk * 32 + 8 * lhalf;
    #pragma unroll
    for (int nt = 0; nt < 8; ++nt) {
      const unsigned short* bp = sW + (nt * 16 + lmod) * LDSTR + kbB;
      FragAB b;
      #pragma unroll
      for (int j = 0; j < 16; ++j) b.u[j] = bp[j < 8 ? j : j + 8];
      acc2[nt] = __builtin_amdgcn_wmma_f32_16x16x32_bf16(
          false, a.b, false, b.b, (short)0, acc2[nt], false, false);
    }
  }

  if (valid) {
    #pragma unroll
    for (int nt = 0; nt < 8; ++nt) {
      float bias = Bb[nt * 16 + lmod];
      #pragma unroll
      for (int r = 0; r < 8; ++r)
        Out[(size_t)(rowBase + r + 8 * lhalf) * HIDC + nt * 16 + lmod] = acc2[nt][r] + bias;
    }
  }
}

// ---------------------------------------------------------------------------
// helpers
// ---------------------------------------------------------------------------
__global__ void wconv_kernel(const float* __restrict__ W, unsigned short* __restrict__ WT) {
  int t = blockIdx.x * blockDim.x + threadIdx.x;      // 128*128 elems
  if (t >= 128 * 128) return;
  int k = t >> 7, n = t & 127;
  WT[n * 128 + k] = f2bf(W[k * 128 + n]);             // transpose + bf16
}

__global__ void copy4_kernel(const float4* __restrict__ in, float4* __restrict__ out, int n4) {
  int i = blockIdx.x * blockDim.x + threadIdx.x;
  if (i < n4) out[i] = in[i];
}

__global__ void zero_kernel(float* __restrict__ p, int n) {
  int i = blockIdx.x * blockDim.x + threadIdx.x;
  if (i < n) p[i] = 0.0f;
}

// one wave per edge: coalesced 512B row gather, native f32 atomic scatter
__global__ void scatter_kernel(const float* __restrict__ X, const int* __restrict__ ei,
                               float* __restrict__ Agg, int nedges) {
  int t = blockIdx.x * blockDim.x + threadIdx.x;
  int e = t >> 5;
  if (e >= nedges) return;
  int lane = t & 31;
  int s = ei[e], d = ei[nedges + e];
  const float4 v = *(const float4*)(X + (size_t)s * HIDC + lane * 4);
  float* dp = Agg + (size_t)d * HIDC + lane * 4;
  unsafeAtomicAdd(dp + 0, v.x);
  unsafeAtomicAdd(dp + 1, v.y);
  unsafeAtomicAdd(dp + 2, v.z);
  unsafeAtomicAdd(dp + 3, v.w);
}

// one wave per node: mean-pool accumulation
__global__ void pool_kernel(const float* __restrict__ H, const int* __restrict__ batch,
                            float* __restrict__ sums, float* __restrict__ cnt, int nnodes) {
  int t = blockIdx.x * blockDim.x + threadIdx.x;
  int i = t >> 5;
  if (i >= nnodes) return;
  int lane = t & 31;
  int g = batch[i];
  const float4 v = *(const float4*)(H + (size_t)i * HIDC + lane * 4);
  float* sp = sums + (size_t)g * HIDC + lane * 4;
  unsafeAtomicAdd(sp + 0, v.x);
  unsafeAtomicAdd(sp + 1, v.y);
  unsafeAtomicAdd(sp + 2, v.z);
  unsafeAtomicAdd(sp + 3, v.w);
  if (lane == 0) unsafeAtomicAdd(cnt + g, 1.0f);
}

// out[g][o] = (sums[g]/max(cnt,1)) @ Wl + bl     (256x64, exact fp32)
__global__ void final_kernel(const float* __restrict__ sums, const float* __restrict__ cnt,
                             const float* __restrict__ Wl, const float* __restrict__ bl,
                             float* __restrict__ out) {
  int g = blockIdx.x;
  int o = threadIdx.x;                                 // 64 threads
  float inv = 1.0f / fmaxf(cnt[g], 1.0f);
  float acc = 0.0f;
  for (int k = 0; k < HIDC; ++k)
    acc += sums[g * HIDC + k] * Wl[k * 64 + o];
  out[g * 64 + o] = acc * inv + bl[o];
}

// ---------------------------------------------------------------------------
extern "C" void kernel_launch(void* const* d_in, const int* in_sizes, int n_in,
                              void* d_out, int out_size, void* d_ws, size_t ws_size,
                              hipStream_t stream) {
  const float* x   = (const float*)d_in[0];
  const int*   ei  = (const int*)  d_in[1];
  const int*   bat = (const int*)  d_in[2];
  const float* W1a = (const float*)d_in[3];  const float* b1a = (const float*)d_in[4];
  const float* W1b = (const float*)d_in[5];  const float* b1b = (const float*)d_in[6];
  const float* W2a = (const float*)d_in[7];  const float* b2a = (const float*)d_in[8];
  const float* W2b = (const float*)d_in[9];  const float* b2b = (const float*)d_in[10];
  const float* Wl  = (const float*)d_in[11]; const float* bl  = (const float*)d_in[12];
  float* out = (float*)d_out;

  const int N = in_sizes[0] / HIDC;   // 50000
  const int E = in_sizes[1] / 2;      // 800000
  const int G = 256;

  // workspace carve
  char* ws = (char*)d_ws;
  size_t off = 0;
  auto carve = [&](size_t bytes) -> void* {
    void* p = ws + off;
    off += (bytes + 255) & ~(size_t)255;
    return p;
  };
  float*          bufA  = (float*)carve((size_t)N * HIDC * 4);
  float*          bufB  = (float*)carve((size_t)N * HIDC * 4);
  unsigned short* WT1a  = (unsigned short*)carve(128 * 128 * 2);
  unsigned short* WT1b  = (unsigned short*)carve(128 * 128 * 2);
  unsigned short* WT2a  = (unsigned short*)carve(128 * 128 * 2);
  unsigned short* WT2b  = (unsigned short*)carve(128 * 128 * 2);
  float*          sums  = (float*)carve((size_t)G * HIDC * 4 + G * 4);
  float*          cnt   = sums + (size_t)G * HIDC;
  (void)n_in; (void)out_size; (void)ws_size;

  const int wgrid   = (128 * 128 + 255) / 256;
  const int n4      = N * (HIDC / 4);
  const int cgrid   = (n4 + 255) / 256;
  const int sgrid   = (E * 32 + 255) / 256;
  const int strips  = (N + 15) / 16;
  const int mgrid   = (strips + 3) / 4;
  const int pgrid   = (N * 32 + 255) / 256;
  const int zn      = G * HIDC + G;
  const int zgrid   = (zn + 255) / 256;

  // 0. weight transpose + bf16 convert
  wconv_kernel<<<wgrid, 256, 0, stream>>>(W1a, WT1a);
  wconv_kernel<<<wgrid, 256, 0, stream>>>(W1b, WT1b);
  wconv_kernel<<<wgrid, 256, 0, stream>>>(W2a, WT2a);
  wconv_kernel<<<wgrid, 256, 0, stream>>>(W2b, WT2b);

  // 1. layer 1: agg = x + scatter(x) ; h1 = MLP1(agg)
  copy4_kernel<<<cgrid, 256, 0, stream>>>((const float4*)x, (float4*)bufA, n4);
  scatter_kernel<<<sgrid, 256, 0, stream>>>(x, ei, bufA, E);
  mlp_kernel<<<mgrid, 128, 0, stream>>>(bufA, WT1a, b1a, WT1b, b1b, bufB, N);

  // 2. layer 2: agg = h1 + scatter(h1) ; h2 = MLP2(agg)
  copy4_kernel<<<cgrid, 256, 0, stream>>>((const float4*)bufB, (float4*)bufA, n4);
  scatter_kernel<<<sgrid, 256, 0, stream>>>(bufB, ei, bufA, E);
  mlp_kernel<<<mgrid, 128, 0, stream>>>(bufA, WT2a, b2a, WT2b, b2b, bufB, N);

  // 3. mean pool + readout
  zero_kernel<<<zgrid, 256, 0, stream>>>(sums, zn);
  pool_kernel<<<pgrid, 256, 0, stream>>>(bufB, bat, sums, cnt, N);
  final_kernel<<<G, 64, 0, stream>>>(sums, cnt, Wl, bl, out);
}
